// MoECooked_38886633898786
// MI455X (gfx1250) — matrix operand order repaired and checked by reading, not compiled, not run on hardware
//
#include <hip/hip_runtime.h>
#include <hip/hip_bf16.h>

#define TKN   8192      // b*l tokens
#define HD    1024      // h
#define HHD   4096      // 4h
#define NE    8         // experts
#define TAUF  0.5f
#define MT    32        // tokens per MLP tile/block

typedef __bf16 bf16x16 __attribute__((ext_vector_type(16)));
typedef float  f32x8   __attribute__((ext_vector_type(8)));

union BFrag { uint4 u[2]; bf16x16 v; };

// ---------------- workspace layout (bytes) ----------------
#define OFF_COUNTS  0
#define OFF_BUCKET  1024
#define OFF_XB      (OFF_BUCKET + (size_t)NE * TKN * 4)                 // bf16 x  [T][HD]
#define OFF_WFCT    (OFF_XB + (size_t)TKN * HD * 2)                     // bf16 WfcT  [E][HHD][HD]
#define OFF_WPROJT  (OFF_WFCT + (size_t)NE * HD * HHD * 2)              // bf16 WprojT[E][HD][HHD]

// ---------------- kernel 1: zero counters ----------------
__global__ void moe_zero_kernel(int* counts) {
    if (threadIdx.x < NE) counts[threadIdx.x] = 0;
}

// ---------------- kernel 2: gating (one wave32 per token) ----------------
__global__ __launch_bounds__(256) void moe_gate_kernel(
    const float* __restrict__ x, const float* __restrict__ gu,
    const float* __restrict__ Wg, float* __restrict__ logits_out,
    __bf16* __restrict__ Xb, int* __restrict__ counts, int* __restrict__ bucket)
{
    int wave = threadIdx.x >> 5, lane = threadIdx.x & 31;
    int t = blockIdx.x * 8 + wave;
    if (t >= TKN) return;

    const float* xr = x + (size_t)t * HD;
    float acc[NE];
#pragma unroll
    for (int e = 0; e < NE; ++e) acc[e] = 0.f;

    for (int j = 0; j < HD / 32; ++j) {
        int i = lane + 32 * j;
        float xv = xr[i];
        Xb[(size_t)t * HD + i] = (__bf16)xv;       // bf16 copy of activations
        const float* wr = Wg + (size_t)i * NE;
#pragma unroll
        for (int e = 0; e < NE; ++e) acc[e] += xv * wr[e];
    }
#pragma unroll
    for (int off = 16; off > 0; off >>= 1) {
#pragma unroll
        for (int e = 0; e < NE; ++e) acc[e] += __shfl_xor(acc[e], off, 32);
    }
    if (lane == 0) {
        int best = 0; float bestz = -3.0e38f;
#pragma unroll
        for (int e = 0; e < NE; ++e) {
            logits_out[(size_t)t * NE + e] = acc[e];
            float u = gu[(size_t)t * NE + e];
            u = fminf(fmaxf(u, 1e-9f), 1.f - 1e-9f);
            float g = -__logf(-__logf(u));
            float z = (acc[e] + g) * (1.0f / TAUF); // monotonic; argmax == ST hard one-hot
            if (z > bestz) { bestz = z; best = e; }
        }
        int pos = atomicAdd(&counts[best], 1);
        bucket[(size_t)best * TKN + pos] = t;       // combine weight is exactly 1.0
    }
}

// ---------------- kernel 3: f32 [R][C] -> bf16 transposed [C][R], per expert slice ----------------
__global__ __launch_bounds__(256) void moe_transpose_kernel(
    const float* __restrict__ src, __bf16* __restrict__ dst, int R, int C)
{
    __shared__ float tile[32][33];
    size_t slice = (size_t)blockIdx.z * R * C;
    const float* s = src + slice;
    __bf16* d = dst + slice;
    int c0 = blockIdx.x * 32, r0 = blockIdx.y * 32;
#pragma unroll
    for (int i = threadIdx.y; i < 32; i += 8)
        tile[i][threadIdx.x] = s[(size_t)(r0 + i) * C + c0 + threadIdx.x];
    __syncthreads();
#pragma unroll
    for (int i = threadIdx.y; i < 32; i += 8)
        d[(size_t)(c0 + i) * R + r0 + threadIdx.x] = (__bf16)tile[threadIdx.x][i];
}

// ---------------- kernel 4: grouped expert MLP via bf16 WMMA ----------------
// block = 256 threads (8 waves), 32 tokens/block.
// Dynamic LDS: Xs [32][HD] bf16 (64KB) + Hs [32][HHD] bf16 (256KB) = 320KB (full WGP LDS)
__device__ __forceinline__ f32x8 wmma_bf16(const BFrag& a, const BFrag& b, f32x8 c) {
    return __builtin_amdgcn_wmma_f32_16x16x32_bf16(false, a.v, false, b.v, (short)0, c, false, false);
}

__global__ __launch_bounds__(256) void moe_mlp_kernel(
    const __bf16* __restrict__ Xb, const __bf16* __restrict__ WfcT,
    const __bf16* __restrict__ WprojT, const int* __restrict__ counts,
    const int* __restrict__ bucket, float* __restrict__ out)
{
    extern __shared__ char smem[];
    __bf16* Xs = (__bf16*)smem;                        // [MT][HD]
    __bf16* Hs = (__bf16*)(smem + (size_t)MT * HD * 2); // [MT][HHD]

    int e    = blockIdx.x / (TKN / MT);
    int tile = blockIdx.x % (TKN / MT);
    int cnt  = counts[e];
    int base = tile * MT;
    if (base >= cnt) return;                      // uniform across block
    int nvalid = min(MT, cnt - base);
    const int* bk = bucket + (size_t)e * TKN + base;

    // ---- gather 32 token rows (bf16) into LDS ----
    {
        int r   = threadIdx.x >> 3;               // 32 rows, 8 threads per row
        int sub = threadIdx.x & 7;
        int row = bk[min(r, nvalid - 1)];
        const uint4* src = (const uint4*)(Xb + (size_t)row * HD);
        uint4*       dst = (uint4*)(Xs + (size_t)r * HD);
#pragma unroll
        for (int i = 0; i < 16; ++i) dst[sub + 8 * i] = src[sub + 8 * i];
    }
    __syncthreads();

    int wave  = threadIdx.x >> 5, lane = threadIdx.x & 31;
    int lhalf = lane >> 4, lmod = lane & 15;      // B-frag: lane%16 = column, lane/16 = K offset 8
    int kAoff = 8 * lhalf;                        // A-frag K base offset per ISA layout

    // ---- stage 1: Hs = relu(Xs[32,HD] @ Wfc[e])^2 ; wave owns 32 N-tiles ----
    const __bf16* We  = WfcT + (size_t)e * HHD * HD;     // [HHD][HD] n-major
    const __bf16* a0p = Xs + (size_t)lmod * HD;          // A rows, M-tile 0 (LDS)
    const __bf16* a1p = Xs + (size_t)(16 + lmod) * HD;   // A rows, M-tile 1 (LDS)
    for (int nt = 0; nt < 32; ++nt) {
        asm volatile("" ::: "memory");            // LICM fence: keep frag loads inside this loop
        int n0 = (wave * 32 + nt) * 16;
        const __bf16* bcol = We + (size_t)(n0 + lmod) * HD + 8 * lhalf;
        __builtin_prefetch(bcol + HD * 16, 0, 1); // next N-tile column (L2)
        f32x8 acc0 = {0.f,0.f,0.f,0.f,0.f,0.f,0.f,0.f};
        f32x8 acc1 = {0.f,0.f,0.f,0.f,0.f,0.f,0.f,0.f};
#pragma unroll 8
        for (int k0 = 0; k0 < HD; k0 += 32) {
            BFrag a0, a1, b;
            a0.u[0] = *(const uint4*)(a0p + k0 + kAoff);
            a0.u[1] = *(const uint4*)(a0p + k0 + kAoff + 16);
            a1.u[0] = *(const uint4*)(a1p + k0 + kAoff);
            a1.u[1] = *(const uint4*)(a1p + k0 + kAoff + 16);
            b.u[0]  = *(const uint4*)(bcol + k0);
            b.u[1]  = *(const uint4*)(bcol + k0 + 16);
            acc0 = wmma_bf16(a0, b, acc0);
            acc1 = wmma_bf16(a1, b, acc1);
        }
#pragma unroll
        for (int v = 0; v < 8; ++v) {             // C layout: lane%16=N, VGPRv -> M=v+8*(lane/16)
            int m = v + 8 * lhalf;
            float r0 = fmaxf(acc0[v], 0.f);
            float r1 = fmaxf(acc1[v], 0.f);
            Hs[(size_t)m * HHD + n0 + lmod]        = (__bf16)(r0 * r0);
            Hs[(size_t)(16 + m) * HHD + n0 + lmod] = (__bf16)(r1 * r1);
        }
    }
    __syncthreads();

    // ---- stage 2: out = Hs[32,HHD] @ Wproj[e] ; wave owns 8 N-tiles ----
    const __bf16* Wp = WprojT + (size_t)e * HD * HHD;    // [HD][HHD] n-major
    const __bf16* h0 = Hs + (size_t)lmod * HHD;
    const __bf16* h1 = Hs + (size_t)(16 + lmod) * HHD;

    int rows0[8], rows1[8];                        // output row ids for this lane's C elements
#pragma unroll
    for (int v = 0; v < 8; ++v) {
        int m = v + 8 * lhalf;
        rows0[v] = bk[min(m, nvalid - 1)];
        rows1[v] = bk[min(16 + m, nvalid - 1)];
    }

    for (int nt = 0; nt < 8; ++nt) {
        asm volatile("" ::: "memory");            // LICM fence
        int n0 = (wave * 8 + nt) * 16;
        const __bf16* bcol = Wp + (size_t)(n0 + lmod) * HHD + 8 * lhalf;
        f32x8 acc0 = {0.f,0.f,0.f,0.f,0.f,0.f,0.f,0.f};
        f32x8 acc1 = {0.f,0.f,0.f,0.f,0.f,0.f,0.f,0.f};
#pragma unroll 4
        for (int k0 = 0; k0 < HHD; k0 += 32) {
            BFrag a0, a1, b;
            a0.u[0] = *(const uint4*)(h0 + k0 + kAoff);
            a0.u[1] = *(const uint4*)(h0 + k0 + kAoff + 16);
            a1.u[0] = *(const uint4*)(h1 + k0 + kAoff);
            a1.u[1] = *(const uint4*)(h1 + k0 + kAoff + 16);
            b.u[0]  = *(const uint4*)(bcol + k0);
            b.u[1]  = *(const uint4*)(bcol + k0 + 16);
            acc0 = wmma_bf16(a0, b, acc0);
            acc1 = wmma_bf16(a1, b, acc1);
        }
#pragma unroll
        for (int v = 0; v < 8; ++v) {
            int m = v + 8 * lhalf;
            if (m < nvalid)      out[(size_t)rows0[v] * HD + n0 + lmod] = acc0[v];
            if (16 + m < nvalid) out[(size_t)rows1[v] * HD + n0 + lmod] = acc1[v];
        }
    }
}

// ---------------- host ----------------
extern "C" void kernel_launch(void* const* d_in, const int* in_sizes, int n_in,
                              void* d_out, int out_size, void* d_ws, size_t ws_size,
                              hipStream_t stream) {
    const float* x     = (const float*)d_in[0];   // [4,2048,1024]
    const float* gu    = (const float*)d_in[1];   // [8192,8]
    const float* Wg    = (const float*)d_in[2];   // [1024,8]
    const float* Wfc   = (const float*)d_in[3];   // [8,1024,4096]
    const float* Wproj = (const float*)d_in[4];   // [8,4096,1024]

    float* out    = (float*)d_out;                // [8192,1024]
    float* logits = out + (size_t)TKN * HD;       // [8192,8]

    char* ws = (char*)d_ws;                       // needs ~145 MB
    int*    counts = (int*)(ws + OFF_COUNTS);
    int*    bucket = (int*)(ws + OFF_BUCKET);
    __bf16* Xb     = (__bf16*)(ws + OFF_XB);
    __bf16* WfcT   = (__bf16*)(ws + OFF_WFCT);
    __bf16* WprojT = (__bf16*)(ws + OFF_WPROJT);

    moe_zero_kernel<<<1, 64, 0, stream>>>(counts);

    moe_gate_kernel<<<TKN / 8, 256, 0, stream>>>(x, gu, Wg, logits, Xb, counts, bucket);

    // Wfc[e]: [HD][HHD] -> WfcT[e]: [HHD][HD]
    moe_transpose_kernel<<<dim3(HHD / 32, HD / 32, NE), dim3(32, 8), 0, stream>>>(Wfc, WfcT, HD, HHD);
    // Wproj[e]: [HHD][HD] -> WprojT[e]: [HD][HHD]
    moe_transpose_kernel<<<dim3(HD / 32, HHD / 32, NE), dim3(32, 8), 0, stream>>>(Wproj, WprojT, HHD, HD);

    size_t smem = (size_t)MT * HD * 2 + (size_t)MT * HHD * 2;  // 64KB + 256KB = 320KB dynamic LDS
    moe_mlp_kernel<<<NE * (TKN / MT), 256, smem, stream>>>(Xb, WfcT, WprojT, counts, bucket, out);
}
